// SlotAttention_86964497810065
// MI455X (gfx1250) — matrix operand order repaired and checked by reading, not compiled, not run on hardware
//
#include <hip/hip_runtime.h>

typedef __attribute__((ext_vector_type(16))) _Float16 v16h;
typedef __attribute__((ext_vector_type(8)))  _Float16 v8h;
typedef __attribute__((ext_vector_type(8)))  float    v8f;

#define LN_EPS 1e-5f
#define ATTN_EPS 1e-6f

constexpr int Bb = 64, Nn = 4096, Dd = 256, Ss = 8, Mhid = 512;
constexpr int RS = Bb * Ss; // 512 rows of slot-space matrices

// ---------------- WMMA helpers (f16 in, f32 acc) ----------------

__device__ inline v8f vzero8f() {
    v8f z = {0.f,0.f,0.f,0.f,0.f,0.f,0.f,0.f};
    return z;
}
__device__ inline v16h vzero16h() {
    v16h z = {(_Float16)0.f,(_Float16)0.f,(_Float16)0.f,(_Float16)0.f,
              (_Float16)0.f,(_Float16)0.f,(_Float16)0.f,(_Float16)0.f,
              (_Float16)0.f,(_Float16)0.f,(_Float16)0.f,(_Float16)0.f,
              (_Float16)0.f,(_Float16)0.f,(_Float16)0.f,(_Float16)0.f};
    return z;
}

// A-matrix 16x32 fragment from row-major [.,K] storage.
// Lane layout: lanes<16: halves0-15 = K(koff+0..15)
//              lanes>=16: halves0-15 = K(koff+16..31).
__device__ inline v16h load_a_frag(const _Float16* rowbase, int koff, int hi) {
    const int o = koff + (hi ? 16 : 0);
    v8h lo = *(const v8h*)(rowbase + o);
    v8h hb = *(const v8h*)(rowbase + o + 8);
    return __builtin_shufflevector(lo, hb, 0,1,2,3,4,5,6,7,8,9,10,11,12,13,14,15);
}
// B-matrix 32x16 fragment: per lane 16 contiguous halves along K (col fixed = lane%16,
// K chunk selected by lane/16). p must already point at col_row + 16*(lane/16) + koff.
__device__ inline v16h load_b_frag(const _Float16* p) {
    v8h lo = *(const v8h*)p;
    v8h hb = *(const v8h*)(p + 8);
    return __builtin_shufflevector(lo, hb, 0,1,2,3,4,5,6,7,8,9,10,11,12,13,14,15);
}
__device__ inline v8f wmma_f16(v16h a, v16h b, v8f c) {
    return __builtin_amdgcn_wmma_f32_16x16x32_f16(false, a, false, b, (short)0, c, false, false);
}

// ---------------- small utility kernels ----------------

__global__ void k_f32_to_f16(const float* __restrict__ s, _Float16* __restrict__ d, int n) {
    int i = blockIdx.x * blockDim.x + threadIdx.x;
    if (i < n) d[i] = (_Float16)s[i];
}
__global__ void k_zero_f32(float* p, int n) {
    int i = blockIdx.x * blockDim.x + threadIdx.x;
    if (i < n) p[i] = 0.f;
}
__global__ void k_init_slots(const float* __restrict__ s0, float* __restrict__ sf,
                             _Float16* __restrict__ sh, int n) {
    int i = blockIdx.x * blockDim.x + threadIdx.x;
    if (i < n) { float v = s0[i]; sf[i] = v; sh[i] = (_Float16)v; }
}

// Row LayerNorm over 256 cols: one wave per row, fp32 in -> f16 out.
__global__ void __launch_bounds__(256)
k_rowln(const float* __restrict__ src, const float* __restrict__ g,
        const float* __restrict__ bb, _Float16* __restrict__ dst, int rows) {
    int lane = threadIdx.x & 31;
    int wave = threadIdx.x >> 5;
    int row  = blockIdx.x * 8 + wave;
    if (row >= rows) return;
    const float* rp = src + row * 256 + lane * 8;
    float x[8];
    float s = 0.f, q = 0.f;
    #pragma unroll
    for (int i = 0; i < 8; ++i) { x[i] = rp[i]; s += x[i]; q += x[i] * x[i]; }
    #pragma unroll
    for (int off = 16; off >= 1; off >>= 1) {
        s += __shfl_xor(s, off, 32);
        q += __shfl_xor(q, off, 32);
    }
    float mean = s * (1.f / 256.f);
    float var  = q * (1.f / 256.f) - mean * mean;
    float inv  = rsqrtf(var + LN_EPS);
    _Float16* dp = dst + row * 256 + lane * 8;
    #pragma unroll
    for (int i = 0; i < 8; ++i) {
        int c = lane * 8 + i;
        dp[i] = (_Float16)((x[i] - mean) * inv * g[c] + bb[c]);
    }
}

// ---------------- fused input-LN + K/V projection ----------------
// Per block: 128 tokens. LN(inputs) -> f16 in LDS, then per wave one 16-token tile:
// k[token][d] (row-major) and v_t[d][token] (transposed) via WMMA.
__global__ void __launch_bounds__(256)
k_ln_proj(const float* __restrict__ inp, const float* __restrict__ g,
          const float* __restrict__ bb, const _Float16* __restrict__ wk,
          const _Float16* __restrict__ wv, _Float16* __restrict__ kout,
          _Float16* __restrict__ vt) {
    __shared__ __align__(16) _Float16 xs[128 * 264]; // 264-half row stride (16B aligned)
    const int lane = threadIdx.x & 31;
    const int wave = threadIdx.x >> 5;
    const int T0 = blockIdx.x * 128;          // flat token base over B*N
    const int b  = T0 / Nn;
    const int n0 = T0 % Nn;                   // never straddles a batch (128 | 4096)

    float gl[8], bl[8];
    #pragma unroll
    for (int i = 0; i < 8; ++i) { gl[i] = g[lane * 8 + i]; bl[i] = bb[lane * 8 + i]; }

    for (int rr = 0; rr < 16; ++rr) {
        int row = wave * 16 + rr;
        const float* rp = inp + (T0 + row) * 256 + lane * 8;
        float x[8];
        float s = 0.f, q = 0.f;
        #pragma unroll
        for (int i = 0; i < 8; ++i) { x[i] = rp[i]; s += x[i]; q += x[i] * x[i]; }
        #pragma unroll
        for (int off = 16; off >= 1; off >>= 1) {
            s += __shfl_xor(s, off, 32);
            q += __shfl_xor(q, off, 32);
        }
        float mean = s * (1.f / 256.f);
        float inv  = rsqrtf(q * (1.f / 256.f) - mean * mean + LN_EPS);
        _Float16* xr = xs + row * 264 + lane * 8;
        #pragma unroll
        for (int i = 0; i < 8; ++i)
            xr[i] = (_Float16)((x[i] - mean) * inv * gl[i] + bl[i]);
    }
    __syncthreads();

    const int ml = lane & 15, hi = lane >> 4;
    const _Float16* arow = xs + (wave * 16 + ml) * 264;
    v16h afr[8];
    #pragma unroll
    for (int kb = 0; kb < 8; ++kb) afr[kb] = load_a_frag(arow, kb * 32, hi);

    #pragma unroll
    for (int which = 0; which < 2; ++which) {
        const _Float16* W = which ? wv : wk;
        for (int nt = 0; nt < 16; ++nt) {
            const int col = nt * 16 + ml;
            const _Float16* wrow = W + col * 256 + 16 * hi;
            v8f acc = vzero8f();
            #pragma unroll
            for (int kb = 0; kb < 8; ++kb)
                acc = wmma_f16(afr[kb], load_b_frag(wrow + kb * 32), acc);
            if (which == 0) {
                // k row-major [token][d]; lane column = d, rows M = r + 8*hi
                int tbase = T0 + wave * 16 + 8 * hi;
                #pragma unroll
                for (int r = 0; r < 8; ++r)
                    kout[(tbase + r) * 256 + col] = (_Float16)acc[r];
            } else {
                // v transposed [b][d][token]; 8 consecutive tokens -> one 16B store
                v8h pk;
                #pragma unroll
                for (int r = 0; r < 8; ++r) pk[r] = (_Float16)acc[r];
                int vidx = b * Dd * Nn + col * Nn + n0 + wave * 16 + 8 * hi;
                *(v8h*)(vt + vidx) = pk;
            }
        }
    }
}

// ---------------- dots + slot-softmax + EPS + partial sums ----------------
// Per wave: 16-token tile. D = q(16 slots x 256) * k_tile^T. Lanes 0-15 each hold the
// 8 slot logits of one token in acc[0..7] -> lane-local softmax over slots.
__global__ void __launch_bounds__(128)
k_dots(const _Float16* __restrict__ q, const _Float16* __restrict__ kmat,
       float* __restrict__ attnF, _Float16* __restrict__ attnH,
       float* __restrict__ sums) {
    const int lane = threadIdx.x & 31, wave = threadIdx.x >> 5;
    const int chunks = Nn / 64;
    const int b  = blockIdx.x / chunks;
    const int t0 = (blockIdx.x % chunks) * 64 + wave * 16;
    const int ml = lane & 15, hi = lane >> 4;

    v16h afr[8];
    if (ml < 8) {
        const _Float16* qr = q + (b * 8 + ml) * 256;
        #pragma unroll
        for (int kb = 0; kb < 8; ++kb) afr[kb] = load_a_frag(qr, kb * 32, hi);
    } else {
        #pragma unroll
        for (int kb = 0; kb < 8; ++kb) afr[kb] = vzero16h();
    }

    const int tok = t0 + ml;
    const _Float16* krow = kmat + (b * Nn + tok) * 256 + 16 * hi;
    v8f acc = vzero8f();
    #pragma unroll
    for (int kb = 0; kb < 8; ++kb)
        acc = wmma_f16(afr[kb], load_b_frag(krow + kb * 32), acc);

    float av[8];
    if (hi == 0) {
        float mx = acc[0];
        #pragma unroll
        for (int r = 1; r < 8; ++r) mx = fmaxf(mx, acc[r]);
        float se = 0.f;
        #pragma unroll
        for (int r = 0; r < 8; ++r) { av[r] = expf(acc[r] - mx); se += av[r]; }
        float inv = 1.f / se;
        #pragma unroll
        for (int r = 0; r < 8; ++r) av[r] = av[r] * inv + ATTN_EPS;
        #pragma unroll
        for (int r = 0; r < 8; ++r) attnF[(b * 8 + r) * Nn + tok] = av[r];
    } else {
        #pragma unroll
        for (int r = 0; r < 8; ++r) av[r] = 0.f; // zero-pad slot rows 8..15
    }
    #pragma unroll
    for (int r = 0; r < 8; ++r)
        attnH[(b * 16 + 8 * hi + r) * Nn + tok] = (_Float16)av[r];

    // per-slot partial sums across the 16 tokens of this tile (lanes 0..15)
    #pragma unroll
    for (int r = 0; r < 8; ++r) {
        float v = av[r];
        v += __shfl_xor(v, 8, 32);
        v += __shfl_xor(v, 4, 32);
        v += __shfl_xor(v, 2, 32);
        v += __shfl_xor(v, 1, 32);
        if (lane == 0) atomicAdd(sums + b * 8 + r, v);
    }
}

// ---------------- updates = (attn_u @ v) / rowsum ----------------
// Per wave: one 16-wide d tile, K loop over N=4096.
__global__ void __launch_bounds__(128)
k_updates(const _Float16* __restrict__ attnH, const _Float16* __restrict__ vt,
          const float* __restrict__ sums, float* __restrict__ upF,
          _Float16* __restrict__ upH) {
    const int lane = threadIdx.x & 31, wave = threadIdx.x >> 5;
    const int b  = blockIdx.x >> 2;
    const int d0 = (blockIdx.x & 3) * 64 + wave * 16;
    const int ml = lane & 15, hi = lane >> 4;

    const _Float16* arow = attnH + (b * 16 + ml) * Nn;
    const int d = d0 + ml;
    const _Float16* vrow = vt + (b * Dd + d) * Nn + 16 * hi;
    v8f acc = vzero8f();
    #pragma unroll 4
    for (int kb = 0; kb < Nn; kb += 32)
        acc = wmma_f16(load_a_frag(arow, kb, hi), load_b_frag(vrow + kb), acc);

    if (hi == 0) { // lanes 0..15 hold slots 0..7 (rows M = r)
        #pragma unroll
        for (int r = 0; r < 8; ++r) {
            float v = acc[r] / sums[b * 8 + r];
            int idx = (b * 8 + r) * 256 + d;
            upF[idx] = v;
            upH[idx] = (_Float16)v;
        }
    }
}

// ---------------- generic WMMA GEMM: out = act(A @ W^T * scale + bias) + addend --------
// A: [Mr,K] f16 row-major; W: [Nc,K] f16 row-major. One 16x16 tile per wave.
__global__ void __launch_bounds__(256)
k_gemm(const _Float16* __restrict__ A, const _Float16* __restrict__ W,
       const float* __restrict__ bias, const float* __restrict__ addend,
       float* __restrict__ outF, _Float16* __restrict__ outH,
       int Mr, int Nc, int K, float scale, int relu) {
    const int lane = threadIdx.x & 31, wave = threadIdx.x >> 5;
    const int tile = blockIdx.x * (blockDim.x >> 5) + wave;
    const int tilesN = Nc >> 4;
    if (tile >= (Mr >> 4) * tilesN) return;
    const int mt = tile / tilesN, nt = tile % tilesN;
    const int ml = lane & 15, hi = lane >> 4;

    const _Float16* arow = A + (mt * 16 + ml) * K;
    const int col = nt * 16 + ml;
    const _Float16* wrow = W + col * K + 16 * hi;
    v8f acc = vzero8f();
    for (int kb = 0; kb < K; kb += 32)
        acc = wmma_f16(load_a_frag(arow, kb, hi), load_b_frag(wrow + kb), acc);

    const float bv = bias ? bias[col] : 0.f;
    #pragma unroll
    for (int r = 0; r < 8; ++r) {
        int mrow = mt * 16 + 8 * hi + r;
        float v = acc[r] * scale + bv;
        if (relu) v = fmaxf(v, 0.f);
        int idx = mrow * Nc + col;
        if (addend) v += addend[idx];
        if (outF) outF[idx] = v;
        if (outH) outH[idx] = (_Float16)v;
    }
}

// ---------------- GRU cell elementwise combine ----------------
__global__ void k_gru(const float* __restrict__ gx, const float* __restrict__ gh,
                      float* __restrict__ slots, _Float16* __restrict__ slotsH, int n) {
    int i = blockIdx.x * blockDim.x + threadIdx.x;
    if (i >= n) return;
    int row = i >> 8, col = i & 255;
    int o = row * 768 + col;
    float xr = gx[o], xz = gx[o + 256], xn = gx[o + 512];
    float hr = gh[o], hz = gh[o + 256], hn = gh[o + 512];
    float r = 1.f / (1.f + expf(-(xr + hr)));
    float z = 1.f / (1.f + expf(-(xz + hz)));
    float nn = tanhf(xn + r * hn);
    float h  = slots[i];
    float out = (1.f - z) * nn + z * h;
    slots[i]  = out;
    slotsH[i] = (_Float16)out;
}

// ---------------- finalize: slots copy + normalized attn ----------------
__global__ void k_final(const float* __restrict__ slots, const float* __restrict__ attnF,
                        const float* __restrict__ sums, float* __restrict__ out) {
    int i = blockIdx.x * blockDim.x + threadIdx.x;
    const int ns = RS * Dd;          // 131072
    const int na = Bb * Ss * Nn;     // 2097152
    if (i < ns) out[i] = slots[i];
    else if (i < ns + na) {
        int j = i - ns;
        out[i] = attnF[j] / sums[j / Nn];
    }
}

extern "C" void kernel_launch(void* const* d_in, const int* in_sizes, int n_in,
                              void* d_out, int out_size, void* d_ws, size_t ws_size,
                              hipStream_t stream) {
    (void)in_sizes; (void)n_in; (void)out_size; (void)ws_size;
    const float* inputs    = (const float*)d_in[0];
    const float* slot_init = (const float*)d_in[1];
    const float* w_k       = (const float*)d_in[2];
    const float* w_q       = (const float*)d_in[3];
    const float* w_v       = (const float*)d_in[4];
    const float* gw_ih     = (const float*)d_in[5];
    const float* gw_hh     = (const float*)d_in[6];
    const float* gb_ih     = (const float*)d_in[7];
    const float* gb_hh     = (const float*)d_in[8];
    const float* ln_in_g   = (const float*)d_in[9];
    const float* ln_in_b   = (const float*)d_in[10];
    const float* ln_s_g    = (const float*)d_in[11];
    const float* ln_s_b    = (const float*)d_in[12];
    const float* ln_f_g    = (const float*)d_in[13];
    const float* ln_f_b    = (const float*)d_in[14];
    const float* mw1       = (const float*)d_in[15];
    const float* mb1       = (const float*)d_in[16];
    const float* mw2       = (const float*)d_in[17];
    const float* mb2       = (const float*)d_in[18];

    char* p = (char*)d_ws;
    auto carve = [&](size_t bytes) -> char* {
        char* r = p; p += (bytes + 255) & ~size_t(255); return r;
    };
    _Float16* kf    = (_Float16*)carve((size_t)Bb * Nn * Dd * 2);   // k [B,N,D] f16
    _Float16* vt    = (_Float16*)carve((size_t)Bb * Dd * Nn * 2);   // v^T [B,D,N] f16
    float*    attnF = (float*)   carve((size_t)Bb * 8 * Nn * 4);    // unnormalized attn fp32
    _Float16* attnH = (_Float16*)carve((size_t)Bb * 16 * Nn * 2);   // padded f16 (rows 8..15 = 0)
    float*    sums  = (float*)   carve(Bb * 8 * 4);
    _Float16* qh    = (_Float16*)carve(RS * 256 * 2);
    float*    slots = (float*)   carve(RS * 256 * 4);
    _Float16* slotsH= (_Float16*)carve(RS * 256 * 2);
    _Float16* slnH  = (_Float16*)carve(RS * 256 * 2);
    float*    upF   = (float*)   carve(RS * 256 * 4);
    _Float16* upH   = (_Float16*)carve(RS * 256 * 2);
    float*    gx    = (float*)   carve(RS * 768 * 4);
    float*    gh    = (float*)   carve(RS * 768 * 4);
    _Float16* ffH   = (_Float16*)carve(RS * 256 * 2);
    _Float16* midH  = (_Float16*)carve((size_t)RS * Mhid * 2);
    _Float16* wkH   = (_Float16*)carve(65536 * 2);
    _Float16* wqH   = (_Float16*)carve(65536 * 2);
    _Float16* wvH   = (_Float16*)carve(65536 * 2);
    _Float16* wihH  = (_Float16*)carve(196608 * 2);
    _Float16* whhH  = (_Float16*)carve(196608 * 2);
    _Float16* w1H   = (_Float16*)carve(131072 * 2);
    _Float16* w2H   = (_Float16*)carve(131072 * 2);

    auto cvt = [&](const float* s, _Float16* d, int n) {
        k_f32_to_f16<<<(n + 255) / 256, 256, 0, stream>>>(s, d, n);
    };
    cvt(w_k, wkH, 65536);
    cvt(w_q, wqH, 65536);
    cvt(w_v, wvH, 65536);
    cvt(gw_ih, wihH, 196608);
    cvt(gw_hh, whhH, 196608);
    cvt(mw1, w1H, 131072);
    cvt(mw2, w2H, 131072);

    // Fused LN(inputs) + K/V projections (memory-dominant stage, runs once)
    k_ln_proj<<<Bb * Nn / 128, 256, 0, stream>>>(inputs, ln_in_g, ln_in_b, wkH, wvH, kf, vt);
    k_init_slots<<<RS, 256, 0, stream>>>(slot_init, slots, slotsH, RS * 256);

    const float qscale = 0.0625f; // D^-0.5 = 256^-0.5 folded into q instead of k
    for (int it = 0; it < 3; ++it) {
        k_rowln<<<RS / 8, 256, 0, stream>>>(slots, ln_s_g, ln_s_b, slnH, RS);
        // q = LN(slots) @ w_q^T * D^-0.5 : tiles = (512/16)*(256/16) = 512 -> 64 blocks
        k_gemm<<<64, 256, 0, stream>>>(slnH, wqH, nullptr, nullptr, nullptr, qh,
                                       RS, 256, 256, qscale, 0);
        k_zero_f32<<<2, 256, 0, stream>>>(sums, Bb * 8);
        k_dots<<<Bb * (Nn / 64), 128, 0, stream>>>(qh, kf, attnF, attnH, sums);
        k_updates<<<Bb * 4, 128, 0, stream>>>(attnH, vt, sums, upF, upH);
        // GRU gates: [512,768] = A[512,256] @ W[768,256]^T : 1536 tiles -> 192 blocks
        k_gemm<<<192, 256, 0, stream>>>(upH, wihH, gb_ih, nullptr, gx, nullptr,
                                        RS, 768, 256, 1.f, 0);
        k_gemm<<<192, 256, 0, stream>>>(slotsH, whhH, gb_hh, nullptr, gh, nullptr,
                                        RS, 768, 256, 1.f, 0);
        k_gru<<<RS, 256, 0, stream>>>(gx, gh, slots, slotsH, RS * 256);
        k_rowln<<<RS / 8, 256, 0, stream>>>(slots, ln_f_g, ln_f_b, ffH, RS);
        // MLP: relu(ff @ w1^T + b1) -> [512,512], then + w2^T + b2 + residual
        k_gemm<<<128, 256, 0, stream>>>(ffH, w1H, mb1, nullptr, nullptr, midH,
                                        RS, Mhid, 256, 1.f, 1);
        k_gemm<<<64, 256, 0, stream>>>(midH, w2H, mb2, slots, slots, slotsH,
                                       RS, 256, Mhid, 1.f, 0);
    }
    const int total = RS * Dd + Bb * Ss * Nn; // 2228224
    k_final<<<(total + 255) / 256, 256, 0, stream>>>(slots, attnF, sums, (float*)d_out);
}
